// EdgeFeature_70231305224240
// MI455X (gfx1250) — matrix operand order repaired and checked by reading, not compiled, not run on hardware
//
#include <hip/hip_runtime.h>
#include <hip/hip_bf16.h>

#define B_ 2
#define L_ 2048
#define K_ 30
#define H_ 128

typedef __attribute__((ext_vector_type(16))) __bf16 v16bf;
typedef __attribute__((ext_vector_type(8)))  float  v8f;

// ---------------- small vector helpers ----------------
__device__ __forceinline__ void normv(float* x) {
  float n = sqrtf(x[0]*x[0] + x[1]*x[1] + x[2]*x[2]);
  n = fmaxf(n, 1e-12f);
  float inv = 1.f / n;
  x[0] *= inv; x[1] *= inv; x[2] *= inv;
}
__device__ __forceinline__ void crossv(const float* a, const float* b, float* c) {
  c[0] = a[1]*b[2] - a[2]*b[1];
  c[1] = a[2]*b[0] - a[0]*b[2];
  c[2] = a[0]*b[1] - a[1]*b[0];
}
__device__ __forceinline__ float dotv(const float* a, const float* b) {
  return a[0]*b[0] + a[1]*b[1] + a[2]*b[2];
}
__device__ __forceinline__ float signf_(float x) {
  return (x > 0.f) ? 1.f : ((x < 0.f) ? -1.f : 0.f);
}
__device__ __forceinline__ float rbf1(float d, int kk) {
  float mu = (20.f / 15.f) * (float)kk;
  float z = (d - mu) * (1.f / 1.25f);
  return __expf(-z * z);
}

// ---------------- last-valid-index per batch ----------------
__global__ void last_kernel(const float* __restrict__ mask, int* __restrict__ last) {
  int b = threadIdx.x;
  if (b < B_) {
    float s = 0.f;
    for (int l = 0; l < L_; ++l) s += mask[b * L_ + l];
    last[b] = (int)s - 1;
  }
}

// ---------------- weight prep: bf16 hi/lo fragments in WMMA per-lane layout ----------------
// layout: frag[((s*32 + lane)*8 + t)*16 + e], lane<16: K=32s+e..  lane>=16: K=32s+16+e
__global__ void prep_w_kernel(const float* __restrict__ w, int Kreal, int ksteps,
                              unsigned short* __restrict__ hi, unsigned short* __restrict__ lo) {
  int idx = blockIdx.x * 256 + threadIdx.x;
  int total = ksteps * 32 * 8 * 16;
  if (idx >= total) return;
  int e = idx & 15;
  int t = (idx >> 4) & 7;
  int l = (idx >> 7) & 31;
  int s = idx >> 12;
  int kbB = (l < 16) ? 0 : 16;
  int k = 32 * s + kbB + e;
  int n = t * 16 + (l & 15);
  float v = (k < Kreal) ? w[k * H_ + n] : 0.f;
  __bf16 h = (__bf16)v;
  __bf16 r = (__bf16)(v - (float)h);
  hi[idx] = __builtin_bit_cast(unsigned short, h);
  lo[idx] = __builtin_bit_cast(unsigned short, r);
}

// ---------------- top-K neighbor selection: one wave32 per row ----------------
__global__ void topk_kernel(const float* __restrict__ xyz, int* __restrict__ eidx) {
  __shared__ float ca[L_ * 3];
  int tid = threadIdx.x;
  int i0 = blockIdx.x * 8;          // 8 rows (waves) per block, same batch (8 | 2048)
  int b = i0 / L_;
  for (int j = tid; j < L_; j += 256) {
    const float* p = xyz + ((size_t)(b * L_ + j) * 6 + 1) * 3;
    ca[j * 3 + 0] = p[0]; ca[j * 3 + 1] = p[1]; ca[j * 3 + 2] = p[2];
  }
  __syncthreads();
  int wave = tid >> 5, lane = tid & 31;
  int li = (i0 + wave) - b * L_;
  float cx = ca[li * 3 + 0], cy = ca[li * 3 + 1], cz = ca[li * 3 + 2];
  float dist[64];
#pragma unroll
  for (int t = 0; t < 64; ++t) {
    int j = t * 32 + lane;
    float dx = cx - ca[j * 3 + 0];
    float dy = cy - ca[j * 3 + 1];
    float dz = cz - ca[j * 3 + 2];
    dist[t] = sqrtf(dx * dx + dy * dy + dz * dz + 1e-6f);
  }
  unsigned long long taken = 0ull;
  int* out = eidx + (size_t)(b * L_ + li) * K_;
  for (int k = 0; k < K_; ++k) {
    float best = 3.4e38f;
    int bj = 0x7fffffff;
#pragma unroll
    for (int t = 0; t < 64; ++t) {
      if (!((taken >> t) & 1ull)) {
        float d = dist[t];
        if (d < best) { best = d; bj = t * 32 + lane; }
      }
    }
    // wave argmin with index tie-break (matches stable top_k)
    for (int m = 16; m >= 1; m >>= 1) {
      float ov = __shfl_xor(best, m, 32);
      int   oj = __shfl_xor(bj,   m, 32);
      if (ov < best || (ov == best && oj < bj)) { best = ov; bj = oj; }
    }
    if ((bj & 31) == lane) taken |= (1ull << (bj >> 5));
    if (lane == 0) out[k] = bj;
  }
}

// ---------------- per-node geometry: frame, O, geo_node (padded to 288) ----------------
__global__ void node_geo_kernel(const float* __restrict__ xyz, const int* __restrict__ last,
                                float* __restrict__ geo, float* __restrict__ frameOut,
                                float* __restrict__ Oout) {
  int idx = blockIdx.x * 256 + threadIdx.x;
  if (idx >= B_ * L_) return;
  int b = idx / L_, l = idx - b * L_;
  const float* at = xyz + (size_t)idx * 18;
  const float* Xb = xyz + (size_t)b * L_ * 18;
  float Ca[3] = { at[3], at[4], at[5] };

  // frame: columns [b, n, b x n]
  float u[3], v[3];
  for (int c = 0; c < 3; ++c) { u[c] = at[c] - Ca[c]; v[c] = Ca[c] - at[6 + c]; }
  normv(u); normv(v);
  float bv[3] = { u[0]-v[0], u[1]-v[1], u[2]-v[2] }; normv(bv);
  float nv[3]; crossv(u, v, nv); normv(nv);
  float c2[3]; crossv(bv, nv, c2);
  float fr[9];
  for (int c = 0; c < 3; ++c) { fr[c*3+0] = bv[c]; fr[c*3+1] = nv[c]; fr[c*3+2] = c2[c]; }
  for (int c = 0; c < 9; ++c) frameOut[(size_t)idx * 9 + c] = fr[c];

  float* row = geo + (size_t)idx * 288;

  // node_dir: atoms {0,2,3,4,5} relative to Ca, rotated into frame
  const int dai[5] = { 0, 2, 3, 4, 5 };
#pragma unroll
  for (int a = 0; a < 5; ++a) {
    float t3[3];
    for (int c = 0; c < 3; ++c) t3[c] = at[dai[a]*3 + c] - Ca[c];
    normv(t3);
    for (int d = 0; d < 3; ++d)
      row[a*3 + d] = t3[0]*fr[d] + t3[1]*fr[3+d] + t3[2]*fr[6+d];
  }

  // node_angle: dihedral + bond from N-Ca-C chain
  const float eps = 1e-7f;
  float dih[3], bnd[3];
#pragma unroll
  for (int s = 0; s < 3; ++s) {
    int q = 3 * l + s - 1;
    float Dd = 0.f, Db = 0.f;
    if (q >= 0 && q <= 3 * L_ - 4) {
      float P[4][3];
      for (int pp = 0; pp < 4; ++pp) {
        int m = q + pp;
        const float* xp = Xb + (size_t)(m / 3) * 18 + (m % 3) * 3;
        P[pp][0] = xp[0]; P[pp][1] = xp[1]; P[pp][2] = xp[2];
      }
      float u2[3], u1[3], u0[3];
      for (int c = 0; c < 3; ++c) {
        u2[c] = P[1][c] - P[0][c];
        u1[c] = P[2][c] - P[1][c];
        u0[c] = P[3][c] - P[2][c];
      }
      normv(u2); normv(u1); normv(u0);
      float n2[3], n1[3];
      crossv(u2, u1, n2); normv(n2);
      crossv(u1, u0, n1); normv(n1);
      float cosD = fminf(fmaxf(dotv(n2, n1), -1.f + eps), 1.f - eps);
      Dd = signf_(dotv(u2, n1)) * acosf(cosD);
      float cosB = fminf(fmaxf(dotv(u2, u1), -1.f + eps), 1.f - eps);
      Db = acosf(cosB);
    }
    dih[s] = Dd; bnd[s] = Db;
  }
  float zl = (l == last[b]) ? 0.f : 1.f;
#pragma unroll
  for (int s = 0; s < 3; ++s) {
    row[15 + s] = zl * cosf(dih[s]);
    row[18 + s] = zl * sinf(dih[s]);
    row[21 + s] = zl * cosf(bnd[s]);
    row[24 + s] = zl * sinf(bnd[s]);
  }

  // O orientation matrix (rows o1, nc2, o1 x nc2); zero-padded ends
  float Om[9] = {0,0,0,0,0,0,0,0,0};
  if (l >= 1 && l <= L_ - 3) {
    const float* cm = Xb + (size_t)(l - 1) * 18 + 3;
    const float* c0 = Xb + (size_t)l * 18 + 3;
    const float* cp = Xb + (size_t)(l + 1) * 18 + 3;
    float uc2[3] = { c0[0]-cm[0], c0[1]-cm[1], c0[2]-cm[2] };
    float uc1[3] = { cp[0]-c0[0], cp[1]-c0[1], cp[2]-c0[2] };
    normv(uc2); normv(uc1);
    float o1[3] = { uc2[0]-uc1[0], uc2[1]-uc1[1], uc2[2]-uc1[2] }; normv(o1);
    float nc2[3]; crossv(uc2, uc1, nc2); normv(nc2);
    float c3[3]; crossv(o1, nc2, c3);
    for (int c = 0; c < 3; ++c) { Om[c] = o1[c]; Om[3+c] = nc2[c]; Om[6+c] = c3[c]; }
  }
  for (int c = 0; c < 9; ++c) Oout[(size_t)idx * 9 + c] = Om[c];

  // node_rbf: 15 intra-residue atom pair distances x 16 RBFs
  const int r0[15] = {0,0,0,0,0,1,1,1,1,2,2,2,3,3,4};
  const int r1[15] = {1,2,3,4,5,2,3,4,5,3,4,5,4,5,5};
#pragma unroll
  for (int p = 0; p < 15; ++p) {
    float dx = at[r0[p]*3+0] - at[r1[p]*3+0];
    float dy = at[r0[p]*3+1] - at[r1[p]*3+1];
    float dz = at[r0[p]*3+2] - at[r1[p]*3+2];
    float d = sqrtf(dx*dx + dy*dy + dz*dz);
    for (int kk = 0; kk < 16; ++kk) row[27 + p*16 + kk] = rbf1(d, kk);
  }
  for (int p = 267; p < 288; ++p) row[p] = 0.f;  // K padding for WMMA
}

// ---------------- per-edge geometry: geo_edge (padded to 128) ----------------
__global__ void edge_geo_kernel(const float* __restrict__ xyz, const int* __restrict__ eidx,
                                const float* __restrict__ frameIn, const float* __restrict__ Oin,
                                float* __restrict__ geo) {
  int e = blockIdx.x * 256 + threadIdx.x;
  if (e >= B_ * L_ * K_) return;
  int bl = e / K_;
  int b = bl / L_;
  int j = eidx[e];
  const float* ai = xyz + (size_t)bl * 18;
  const float* aj = xyz + (size_t)(b * L_ + j) * 18;
  float Ci[3] = { ai[3], ai[4], ai[5] };
  const float* fr = frameIn + (size_t)bl * 9;
  float* row = geo + (size_t)e * 128;

  // edge_dir: 6 neighbor atoms into i's frame
#pragma unroll
  for (int a = 0; a < 6; ++a) {
    float t3[3] = { aj[a*3+0]-Ci[0], aj[a*3+1]-Ci[1], aj[a*3+2]-Ci[2] };
    normv(t3);
    for (int d = 0; d < 3; ++d)
      row[a*3 + d] = t3[0]*fr[d] + t3[1]*fr[3+d] + t3[2]*fr[6+d];
  }

  // edge_ori: dU + quaternion of Om^T * Oj
  const float* Om = Oin + (size_t)bl * 9;
  const float* Oj = Oin + (size_t)(b * L_ + j) * 9;
  float dX[3] = { aj[3]-Ci[0], aj[4]-Ci[1], aj[5]-Ci[2] };
  float dU[3];
  for (int i3 = 0; i3 < 3; ++i3)
    dU[i3] = Om[i3*3+0]*dX[0] + Om[i3*3+1]*dX[1] + Om[i3*3+2]*dX[2];
  normv(dU);
  row[18] = dU[0]; row[19] = dU[1]; row[20] = dU[2];

  float R[9];
  for (int i3 = 0; i3 < 3; ++i3)
    for (int m3 = 0; m3 < 3; ++m3)
      R[i3*3+m3] = Om[0*3+i3]*Oj[0*3+m3] + Om[1*3+i3]*Oj[1*3+m3] + Om[2*3+i3]*Oj[2*3+m3];
  float Rxx = R[0], Ryy = R[4], Rzz = R[8];
  float q0 = 0.5f * sqrtf(fabsf(1.f + Rxx - Ryy - Rzz)) * signf_(R[7] - R[5]);
  float q1 = 0.5f * sqrtf(fabsf(1.f - Rxx + Ryy - Rzz)) * signf_(R[2] - R[6]);
  float q2 = 0.5f * sqrtf(fabsf(1.f - Rxx - Ryy + Rzz)) * signf_(R[3] - R[1]);
  float qw = 0.5f * sqrtf(fmaxf(0.f, 1.f + Rxx + Ryy + Rzz));
  float qn = fmaxf(sqrtf(q0*q0 + q1*q1 + q2*q2 + qw*qw), 1e-12f);
  row[21] = q0/qn; row[22] = q1/qn; row[23] = q2/qn; row[24] = qw/qn;

  // edge_rbf: 6 neighbor atoms vs Ca_i, 16 RBFs each
#pragma unroll
  for (int a = 0; a < 6; ++a) {
    float dx = aj[a*3+0]-Ci[0], dy = aj[a*3+1]-Ci[1], dz = aj[a*3+2]-Ci[2];
    float d = sqrtf(dx*dx + dy*dy + dz*dz);
    for (int kk = 0; kk < 16; ++kk) row[25 + a*16 + kk] = rbf1(d, kk);
  }
  for (int p = 121; p < 128; ++p) row[p] = 0.f;  // K padding for WMMA
}

// ---------------- fused GEMM (bf16-split WMMA) + bias + LayerNorm ----------------
// Each wave computes 16 rows x 128 channels. 3 WMMAs per k-tile per n-tile.
__global__ void gemm_ln_kernel(const float* __restrict__ in, int Kp,
                               const unsigned short* __restrict__ wf_hi,
                               const unsigned short* __restrict__ wf_lo,
                               const float* __restrict__ bias,
                               const float* __restrict__ g, const float* __restrict__ bo,
                               float* __restrict__ out) {
  int wave = (blockIdx.x * blockDim.x + threadIdx.x) >> 5;
  int lane = threadIdx.x & 31;
  int row0 = wave * 16;
  int ksteps = Kp >> 5;
  int kb    = (lane < 16) ? 0 : 16;  // A fragment K base
  int m_add = (lane < 16) ? 0 : 8;   // C/D row half
  int ncol  = lane & 15;

  v8f acc[8];
#pragma unroll
  for (int t = 0; t < 8; ++t)
#pragma unroll
    for (int r = 0; r < 8; ++r) acc[t][r] = 0.f;

  const float* arow = in + (size_t)(row0 + (lane & 15)) * Kp;
  for (int s = 0; s < ksteps; ++s) {
    const float* ap = arow + 32 * s + kb;
    float4 f0 = *(const float4*)(ap);
    float4 f1 = *(const float4*)(ap + 4);
    float4 f2 = *(const float4*)(ap + 8);
    float4 f3 = *(const float4*)(ap + 12);
    float fa[16] = { f0.x,f0.y,f0.z,f0.w, f1.x,f1.y,f1.z,f1.w,
                     f2.x,f2.y,f2.z,f2.w, f3.x,f3.y,f3.z,f3.w };
    v16bf a_hi, a_lo;
#pragma unroll
    for (int e2 = 0; e2 < 16; ++e2) {
      __bf16 h = (__bf16)fa[e2];
      a_hi[e2] = h;
      a_lo[e2] = (__bf16)(fa[e2] - (float)h);
    }
    size_t fbase = (size_t)(s * 32 + lane) * 128;
#pragma unroll
    for (int t = 0; t < 8; ++t) {
      const uint4* ph = (const uint4*)(wf_hi + fbase + t * 16);
      const uint4* pl = (const uint4*)(wf_lo + fbase + t * 16);
      uint4 h0 = ph[0], h1 = ph[1];
      uint4 l0 = pl[0], l1 = pl[1];
      unsigned int hw[8] = { h0.x,h0.y,h0.z,h0.w,h1.x,h1.y,h1.z,h1.w };
      unsigned int lw[8] = { l0.x,l0.y,l0.z,l0.w,l1.x,l1.y,l1.z,l1.w };
      v16bf b_hi, b_lo;
#pragma unroll
      for (int wd = 0; wd < 8; ++wd) {
        b_hi[2*wd]   = __builtin_bit_cast(__bf16, (unsigned short)(hw[wd] & 0xffffu));
        b_hi[2*wd+1] = __builtin_bit_cast(__bf16, (unsigned short)(hw[wd] >> 16));
        b_lo[2*wd]   = __builtin_bit_cast(__bf16, (unsigned short)(lw[wd] & 0xffffu));
        b_lo[2*wd+1] = __builtin_bit_cast(__bf16, (unsigned short)(lw[wd] >> 16));
      }
      acc[t] = __builtin_amdgcn_wmma_f32_16x16x32_bf16(false, a_hi, false, b_hi, (short)0, acc[t], false, false);
      acc[t] = __builtin_amdgcn_wmma_f32_16x16x32_bf16(false, a_hi, false, b_lo, (short)0, acc[t], false, false);
      acc[t] = __builtin_amdgcn_wmma_f32_16x16x32_bf16(false, a_lo, false, b_hi, (short)0, acc[t], false, false);
    }
  }

  // + bias
#pragma unroll
  for (int t = 0; t < 8; ++t) {
    float bv = bias[t * 16 + ncol];
#pragma unroll
    for (int r = 0; r < 8; ++r) acc[t][r] += bv;
  }

  // LayerNorm per output row (128 channels live across 8 n-tiles x 16 lanes of a half)
#pragma unroll
  for (int r = 0; r < 8; ++r) {
    float s1 = 0.f, s2 = 0.f;
#pragma unroll
    for (int t = 0; t < 8; ++t) { float v = acc[t][r]; s1 += v; s2 += v * v; }
    for (int mm = 1; mm <= 8; mm <<= 1) {
      s1 += __shfl_xor(s1, mm, 32);
      s2 += __shfl_xor(s2, mm, 32);
    }
    float mu  = s1 * (1.f / 128.f);
    float var = s2 * (1.f / 128.f) - mu * mu;
    float inv = rsqrtf(var + 1e-5f);
    float* op = out + (size_t)(row0 + r + m_add) * H_;
#pragma unroll
    for (int t = 0; t < 8; ++t) {
      int n = t * 16 + ncol;
      op[n] = (acc[t][r] - mu) * inv * g[n] + bo[n];
    }
  }
}

extern "C" void kernel_launch(void* const* d_in, const int* in_sizes, int n_in,
                              void* d_out, int out_size, void* d_ws, size_t ws_size,
                              hipStream_t stream) {
  const float* xyz       = (const float*)d_in[0];
  const float* mask      = (const float*)d_in[1];
  const float* node_w    = (const float*)d_in[2];
  const float* node_b    = (const float*)d_in[3];
  const float* edge_w    = (const float*)d_in[4];
  const float* edge_b    = (const float*)d_in[5];
  const float* ln_node_g = (const float*)d_in[6];
  const float* ln_node_b = (const float*)d_in[7];
  const float* ln_edge_g = (const float*)d_in[8];
  const float* ln_edge_b = (const float*)d_in[9];

  // output partition: node (B*L*H), edge (B*L*K*H), E_idx (B*L*K as int32 bits)
  float* node_out = (float*)d_out;
  float* edge_out = node_out + (size_t)B_ * L_ * H_;
  int*   eidx_out = (int*)(edge_out + (size_t)B_ * L_ * K_ * H_);

  // workspace partition (all offsets 256B-aligned)
  char* ws = (char*)d_ws;
  size_t o = 0;
  int*   ws_last  = (int*)(ws + o);            o += 256;
  float* ws_frame = (float*)(ws + o);          o += (size_t)B_ * L_ * 9 * 4;      // 147456
  float* ws_O     = (float*)(ws + o);          o += (size_t)B_ * L_ * 9 * 4;      // 147456
  float* ws_gn    = (float*)(ws + o);          o += (size_t)B_ * L_ * 288 * 4;    // 4718592
  float* ws_ge    = (float*)(ws + o);          o += (size_t)B_ * L_ * K_ * 128 * 4; // 62914560
  unsigned short* nwf_hi = (unsigned short*)(ws + o); o += (size_t)9 * 4096 * 2;  // 73728
  unsigned short* nwf_lo = (unsigned short*)(ws + o); o += (size_t)9 * 4096 * 2;
  unsigned short* ewf_hi = (unsigned short*)(ws + o); o += (size_t)4 * 4096 * 2;  // 32768
  unsigned short* ewf_lo = (unsigned short*)(ws + o); o += (size_t)4 * 4096 * 2;

  last_kernel<<<1, 64, 0, stream>>>(mask, ws_last);
  prep_w_kernel<<<(9 * 4096 + 255) / 256, 256, 0, stream>>>(node_w, 267, 9, nwf_hi, nwf_lo);
  prep_w_kernel<<<(4 * 4096 + 255) / 256, 256, 0, stream>>>(edge_w, 121, 4, ewf_hi, ewf_lo);

  topk_kernel<<<(B_ * L_) / 8, 256, 0, stream>>>(xyz, eidx_out);
  node_geo_kernel<<<(B_ * L_) / 256, 256, 0, stream>>>(xyz, ws_last, ws_gn, ws_frame, ws_O);
  edge_geo_kernel<<<(B_ * L_ * K_) / 256, 256, 0, stream>>>(xyz, eidx_out, ws_frame, ws_O, ws_ge);

  // node GEMM: M=4096 rows -> 256 waves -> 32 blocks of 8 waves
  gemm_ln_kernel<<<(B_ * L_) / 16 / 8, 256, 0, stream>>>(
      ws_gn, 288, nwf_hi, nwf_lo, node_b, ln_node_g, ln_node_b, node_out);
  // edge GEMM: M=122880 rows -> 7680 waves -> 960 blocks
  gemm_ln_kernel<<<(B_ * L_ * K_) / 16 / 8, 256, 0, stream>>>(
      ws_ge, 128, ewf_hi, ewf_lo, edge_b, ln_edge_g, ln_edge_b, edge_out);
}
